// Generator_24670292149162
// MI455X (gfx1250) — matrix operand order repaired and checked by reading, not compiled, not run on hardware
//
#include <hip/hip_runtime.h>
#include <stdint.h>
#include <math.h>

// ---------------------------------------------------------------------------
// Types for CDNA5 WMMA (wave32)
// ---------------------------------------------------------------------------
typedef __attribute__((ext_vector_type(16))) __bf16 v16bf;
typedef __attribute__((ext_vector_type(8)))  __bf16 v8bf;
typedef __attribute__((ext_vector_type(8)))  float  v8f;
typedef __attribute__((ext_vector_type(4)))  float  v4f;

#define B_ROWS 65536
#define EPSBN  1e-5f

// ---------------------------------------------------------------------------
// Fused GEMM:  Y[B,N] = act(A)[B,K] @ W[N,K]^T + bias
//   - WMMA bf16 16x16x32, fp32 accumulate
//   - K, LDY are compile-time (powers of two) -> address math is shifts and
//     the 8 row-strided stores per tile fold into immediate offsets
//   - A_IS_F32: A is fp32 (layer 1 input z), converted to bf16 at load
//   - ACT:      A is bf16 pre-BN activations; apply per-column affine
//               (BatchNorm folded) + LeakyReLU(0.2) at load
//   - STATS:    accumulate per-output-column sum / sum^2 (for next BN)
//   - OUT_F32:  write fp32 (logits) instead of bf16
// One wave computes a 32x(16*NT) tile (MT=2 row-tiles). Block = 4 waves.
// ---------------------------------------------------------------------------
template <int MT, int NT, int K, int LDY, bool A_IS_F32, bool ACT, bool STATS,
          bool OUT_F32>
__global__ __launch_bounds__(128, 2) void gemm_wmma(
    const void* __restrict__ Av, const __bf16* __restrict__ W,
    const float* __restrict__ bias,
    const float* __restrict__ act_scale, const float* __restrict__ act_shift,
    __bf16* __restrict__ Ybf, float* __restrict__ Yf,
    float* __restrict__ colsum, float* __restrict__ colsq) {
  const int wave = threadIdx.x >> 5;
  const int lane = threadIdx.x & 31;
  const int kh   = lane >> 4;   // lane half selects K sub-chunk
  const int r    = lane & 15;   // row (A) / column (B) within tile
  const int m0   = (blockIdx.x * 4 + wave) * (16 * MT);
  const int n0   = blockIdx.y * (16 * NT);

  const float*  Af = (const float*)Av;
  const __bf16* Ab = (const __bf16*)Av;

  v8f c[MT][NT];
#pragma unroll
  for (int mt = 0; mt < MT; ++mt)
#pragma unroll
    for (int t = 0; t < NT; ++t)
#pragma unroll
      for (int e = 0; e < 8; ++e) c[mt][t][e] = 0.0f;

  for (int k = 0; k < K; k += 32) {
    v16bf a[MT];
#pragma unroll
    for (int mt = 0; mt < MT; ++mt) {
      const size_t rowoff = (size_t)(m0 + mt * 16 + r) * (size_t)K;
      const int k0 = k + 8 * kh;        // frag elements 0..7
      const int k1 = k + 16 + 8 * kh;   // frag elements 8..15
      if (mt == 0 && k + 32 < K) {
        // CDNA5 prefetch path (global_prefetch_b8) for next A chunk
        const size_t esz = A_IS_F32 ? 4u : 2u;
        __builtin_prefetch((const char*)Av + (rowoff + (size_t)(k0 + 32)) * esz,
                           0, 1);
      }
      if (A_IS_F32) {
        v4f f0 = *(const v4f*)(Af + rowoff + k0);
        v4f f1 = *(const v4f*)(Af + rowoff + k0 + 4);
        v4f f2 = *(const v4f*)(Af + rowoff + k1);
        v4f f3 = *(const v4f*)(Af + rowoff + k1 + 4);
#pragma unroll
        for (int i = 0; i < 4; ++i) {
          a[mt][i]      = (__bf16)f0[i];
          a[mt][4 + i]  = (__bf16)f1[i];
          a[mt][8 + i]  = (__bf16)f2[i];
          a[mt][12 + i] = (__bf16)f3[i];
        }
      } else if (ACT) {
        v8bf lo = *(const v8bf*)(Ab + rowoff + k0);
        v8bf hi = *(const v8bf*)(Ab + rowoff + k1);
#pragma unroll
        for (int i = 0; i < 8; ++i) {
          float x = (float)lo[i];
          x = fmaf(x, act_scale[k0 + i], act_shift[k0 + i]);
          x = x > 0.0f ? x : 0.2f * x;              // LeakyReLU(0.2)
          a[mt][i] = (__bf16)x;
          float y = (float)hi[i];
          y = fmaf(y, act_scale[k1 + i], act_shift[k1 + i]);
          y = y > 0.0f ? y : 0.2f * y;
          a[mt][8 + i] = (__bf16)y;
        }
      } else {
        v8bf lo = *(const v8bf*)(Ab + rowoff + k0);
        v8bf hi = *(const v8bf*)(Ab + rowoff + k1);
        a[mt] = __builtin_shufflevector(lo, hi, 0, 1, 2, 3, 4, 5, 6, 7, 8, 9,
                                        10, 11, 12, 13, 14, 15);
      }
    }
#pragma unroll
    for (int t = 0; t < NT; ++t) {
      // B fragment: lane r = column n; 8 K at k + 8*kh, 8 K at k + 16 + 8*kh
      const size_t woff =
          (size_t)(n0 + t * 16 + r) * (size_t)K + (size_t)(k + 8 * kh);
      v8bf b0 = *(const v8bf*)(W + woff);
      v8bf b1 = *(const v8bf*)(W + woff + 16);
      v16bf bf = __builtin_shufflevector(b0, b1, 0, 1, 2, 3, 4, 5, 6, 7, 8, 9,
                                         10, 11, 12, 13, 14, 15);
#pragma unroll
      for (int mt = 0; mt < MT; ++mt)
        c[mt][t] = __builtin_amdgcn_wmma_f32_16x16x32_bf16(
            false, a[mt], false, bf, (short)0, c[mt][t], false, false);
    }
  }

  // Epilogue: bias, store (base + constant-offset rows), batch statistics
#pragma unroll
  for (int t = 0; t < NT; ++t) {
    const int n    = n0 + t * 16 + r;  // C layout: lane -> column
    const float bs = bias[n];
#pragma unroll
    for (int mt = 0; mt < MT; ++mt) {
      // C layout: vgpr j + lane-half -> row;  j*LDY is a compile-time
      // constant, so these 8 stores share one address + immediate offsets.
      const size_t base = (size_t)(m0 + mt * 16 + 8 * kh) * (size_t)LDY + n;
      float psum = 0.0f, psq = 0.0f;
#pragma unroll
      for (int j = 0; j < 8; ++j) {
        const float v = c[mt][t][j] + bs;
        if (OUT_F32) Yf[base + (size_t)(j * LDY)] = v;
        else         Ybf[base + (size_t)(j * LDY)] = (__bf16)v;
        psum += v;
        psq  = fmaf(v, v, psq);
      }
      if (STATS) {
        atomicAdd(&colsum[n], psum);
        atomicAdd(&colsq[n], psq);
      }
    }
  }
}

// ---------------------------------------------------------------------------
// Small helper kernels
// ---------------------------------------------------------------------------
__global__ void zero_f32(float* __restrict__ p, int n) {
  int i = blockIdx.x * blockDim.x + threadIdx.x;
  if (i < n) p[i] = 0.0f;
}

__global__ void convert_f32_bf16(const float* __restrict__ src,
                                 __bf16* __restrict__ dst, int n) {
  int i = blockIdx.x * blockDim.x + threadIdx.x;
  if (i < n) dst[i] = (__bf16)src[i];
}

__global__ void pad_w4_kernel(const float* __restrict__ W4,
                              __bf16* __restrict__ dst) {
  int i = blockIdx.x * blockDim.x + threadIdx.x;  // [32,512] padded
  if (i >= 32 * 512) return;
  int row = i >> 9, col = i & 511;
  dst[i] = (row < 25) ? (__bf16)W4[row * 512 + col] : (__bf16)0.0f;
}

__global__ void pad_b4_kernel(const float* __restrict__ b4,
                              float* __restrict__ dst) {
  int i = threadIdx.x;
  if (i < 32) dst[i] = (i < 25) ? b4[i] : 0.0f;
}

// Fold BatchNorm (training mode, biased var) into per-column affine
__global__ void bn_finalize(const float* __restrict__ s,
                            const float* __restrict__ q,
                            const float* __restrict__ g,
                            const float* __restrict__ bt,
                            float* __restrict__ sc, float* __restrict__ sh,
                            int n, float invB) {
  int i = blockIdx.x * blockDim.x + threadIdx.x;
  if (i >= n) return;
  float m = s[i] * invB;
  float v = fmaf(-m, m, q[i] * invB);
  float k = g[i] * rsqrtf(v + EPSBN);
  sc[i] = k;
  sh[i] = fmaf(-m, k, bt[i]);
}

// ---------------------------------------------------------------------------
// Threefry2x32-20 (matches JAX) + masked Gumbel top-15 sampler
// ---------------------------------------------------------------------------
__device__ __forceinline__ uint32_t rotl32(uint32_t x, int r) {
  return (x << r) | (x >> (32 - r));
}

__device__ __forceinline__ void threefry2x32(uint32_t k0, uint32_t k1,
                                             uint32_t x0, uint32_t x1,
                                             uint32_t& o0, uint32_t& o1) {
  const uint32_t k2 = k0 ^ k1 ^ 0x1BD11BDAu;
  x0 += k0; x1 += k1;
#define TF_R4(a, b, c, d)                                   \
  x0 += x1; x1 = rotl32(x1, a); x1 ^= x0;                   \
  x0 += x1; x1 = rotl32(x1, b); x1 ^= x0;                   \
  x0 += x1; x1 = rotl32(x1, c); x1 ^= x0;                   \
  x0 += x1; x1 = rotl32(x1, d); x1 ^= x0;
  TF_R4(13, 15, 26, 6)  x0 += k1; x1 += k2 + 1u;
  TF_R4(17, 29, 16, 24) x0 += k2; x1 += k0 + 2u;
  TF_R4(13, 15, 26, 6)  x0 += k0; x1 += k1 + 3u;
  TF_R4(17, 29, 16, 24) x0 += k1; x1 += k2 + 4u;
  TF_R4(13, 15, 26, 6)  x0 += k2; x1 += k0 + 5u;
#undef TF_R4
  o0 = x0; o1 = x1;
}

// word j of jax.random.split(key(42), 15) flat output (first-half/second-half counter split)
__device__ __forceinline__ uint32_t jax_split_word(uint32_t j) {
  uint32_t a, b;
  if (j < 15u) { threefry2x32(0u, 42u, j, j + 15u, a, b); return a; }
  threefry2x32(0u, 42u, j - 15u, j, a, b);
  return b;
}

__global__ __launch_bounds__(256) void gumbel_topk(
    const float* __restrict__ logits, float* __restrict__ out, int ld) {
  const int row = blockIdx.x * 256 + threadIdx.x;
  float lg[25];
#pragma unroll
  for (int c = 0; c < 25; ++c) lg[c] = logits[(size_t)row * ld + c];

  uint32_t mask = 0x1FFFFFFu;  // 25 ones
  const uint32_t HALF = (uint32_t)B_ROWS * 25u / 2u;  // 819200

  for (int p = 0; p < 15; ++p) {
    const uint32_t key0 = jax_split_word(2u * p);
    const uint32_t key1 = jax_split_word(2u * p + 1u);
    float best = -INFINITY;
    int   bi   = 0;
#pragma unroll
    for (int c = 0; c < 25; ++c) {
      const uint32_t e = (uint32_t)row * 25u + (uint32_t)c;
      uint32_t o0, o1, bits;
      if (e < HALF) { threefry2x32(key0, key1, e, e + HALF, o0, o1); bits = o0; }
      else          { threefry2x32(key0, key1, e - HALF, e, o0, o1); bits = o1; }
      const float u   = __uint_as_float((bits >> 9) | 0x3f800000u) - 1.0f;
      const float gum = -logf(-logf(u));
      const float val = (((mask >> c) & 1u) ? lg[c] : -1e9f) + gum;
      if (val > best) { best = val; bi = c; }
    }
    mask &= ~(1u << bi);
    out[(size_t)row * 15 + p] = (float)bi * (1.0f / 24.0f);
  }
}

// ---------------------------------------------------------------------------
// Launch
// ---------------------------------------------------------------------------
extern "C" void kernel_launch(void* const* d_in, const int* in_sizes, int n_in,
                              void* d_out, int out_size, void* d_ws,
                              size_t ws_size, hipStream_t stream) {
  (void)in_sizes; (void)n_in; (void)out_size; (void)ws_size;
  const float* z   = (const float*)d_in[0];
  const float* W1  = (const float*)d_in[1];
  const float* b1  = (const float*)d_in[2];
  const float* g1  = (const float*)d_in[3];
  const float* bt1 = (const float*)d_in[4];
  const float* W2  = (const float*)d_in[5];
  const float* b2  = (const float*)d_in[6];
  const float* g2  = (const float*)d_in[7];
  const float* bt2 = (const float*)d_in[8];
  const float* W3  = (const float*)d_in[9];
  const float* b3  = (const float*)d_in[10];
  const float* g3  = (const float*)d_in[11];
  const float* bt3 = (const float*)d_in[12];
  const float* W4  = (const float*)d_in[13];
  const float* b4  = (const float*)d_in[14];

  char* ws = (char*)d_ws;
  size_t off = 0;
  auto alloc = [&](size_t bytes) -> char* {
    char* p = ws + off;
    off += (bytes + 255) & ~(size_t)255;
    return p;
  };

  __bf16* w1b = (__bf16*)alloc((size_t)512 * 128 * 2);
  __bf16* w2b = (__bf16*)alloc((size_t)1024 * 512 * 2);
  __bf16* w3b = (__bf16*)alloc((size_t)512 * 1024 * 2);
  __bf16* w4b = (__bf16*)alloc((size_t)32 * 512 * 2);
  float*  b4p = (float*)alloc(32 * 4);
  float*  st  = (float*)alloc(4096 * 4);  // s1 q1 s2 q2 s3 q3
  float *s1 = st, *q1 = st + 512, *s2 = st + 1024, *q2 = st + 2048,
        *s3 = st + 3072, *q3 = st + 3584;
  float* sc1 = (float*)alloc(512 * 4);
  float* sh1 = (float*)alloc(512 * 4);
  float* sc2 = (float*)alloc(1024 * 4);
  float* sh2 = (float*)alloc(1024 * 4);
  float* sc3 = (float*)alloc(512 * 4);
  float* sh3 = (float*)alloc(512 * 4);
  __bf16* h1 = (__bf16*)alloc((size_t)B_ROWS * 512 * 2);
  __bf16* h2 = (__bf16*)alloc((size_t)B_ROWS * 1024 * 2);
  __bf16* h3 = h1;  // h1 dead after GEMM2; reuse for h3
  float* logits = (float*)alloc((size_t)B_ROWS * 32 * 4);

  // Weight conversion / padding / stats reset
  convert_f32_bf16<<<(512 * 128 + 255) / 256, 256, 0, stream>>>(W1, w1b,
                                                                512 * 128);
  convert_f32_bf16<<<(1024 * 512 + 255) / 256, 256, 0, stream>>>(W2, w2b,
                                                                 1024 * 512);
  convert_f32_bf16<<<(512 * 1024 + 255) / 256, 256, 0, stream>>>(W3, w3b,
                                                                 512 * 1024);
  pad_w4_kernel<<<(32 * 512 + 255) / 256, 256, 0, stream>>>(W4, w4b);
  pad_b4_kernel<<<1, 32, 0, stream>>>(b4, b4p);
  zero_f32<<<16, 256, 0, stream>>>(st, 4096);

  const dim3 blk(128);
  const float invB = 1.0f / (float)B_ROWS;
  const int gx = B_ROWS / 128;  // 4 waves * 32 rows per block

  // Layer 1: z(f32) @ W1^T  -> h1 (pre-BN bf16) + stats
  gemm_wmma<2, 4, 128, 512, true, false, true, false>
      <<<dim3(gx, 512 / 64), blk, 0, stream>>>(z, w1b, b1, nullptr, nullptr,
                                               h1, nullptr, s1, q1);
  bn_finalize<<<2, 256, 0, stream>>>(s1, q1, g1, bt1, sc1, sh1, 512, invB);

  // Layer 2: BN+Leaky(h1) @ W2^T -> h2 + stats
  gemm_wmma<2, 4, 512, 1024, false, true, true, false>
      <<<dim3(gx, 1024 / 64), blk, 0, stream>>>(h1, w2b, b2, sc1, sh1, h2,
                                                nullptr, s2, q2);
  bn_finalize<<<4, 256, 0, stream>>>(s2, q2, g2, bt2, sc2, sh2, 1024, invB);

  // Layer 3: BN+Leaky(h2) @ W3^T -> h3 + stats
  gemm_wmma<2, 4, 1024, 512, false, true, true, false>
      <<<dim3(gx, 512 / 64), blk, 0, stream>>>(h2, w3b, b3, sc2, sh2, h3,
                                               nullptr, s3, q3);
  bn_finalize<<<2, 256, 0, stream>>>(s3, q3, g3, bt3, sc3, sh3, 512, invB);

  // Layer 4: BN+Leaky(h3) @ W4^T -> logits (fp32, padded to 32 cols)
  gemm_wmma<2, 2, 512, 32, false, true, false, true>
      <<<dim3(gx, 1), blk, 0, stream>>>(h3, w4b, b4p, sc3, sh3, nullptr,
                                        logits, nullptr, nullptr);

  // Gumbel top-15 sampling (threefry2x32, JAX semantics)
  gumbel_topk<<<B_ROWS / 256, 256, 0, stream>>>(logits, (float*)d_out, 32);
}
